// TransformerConv_59674275611327
// MI455X (gfx1250) — matrix operand order repaired and checked by reading, not compiled, not run on hardware
//
#include <hip/hip_runtime.h>
#include <math.h>

typedef __attribute__((ext_vector_type(16))) _Float16 v16h;
typedef __attribute__((ext_vector_type(8)))  float    v8f;

#define IN_F   128
#define HD     128
#define QKVS_W 512
#define NH     4

// ---- monotonic float<->uint encoding for atomic max on floats ----
static __device__ __forceinline__ unsigned fenc(float f) {
    unsigned u = __float_as_uint(f);
    return (u & 0x80000000u) ? ~u : (u | 0x80000000u);
}
static __device__ __forceinline__ float fdec(unsigned u) {
    return (u & 0x80000000u) ? __uint_as_float(u ^ 0x80000000u)
                             : __uint_as_float(~u);
}

// ---------------------------------------------------------------------------
// Pack Wq|Wk|Wv|Wskip (each [128,128] row-major, k-major) into WMMA B-fragment
// order, fp16. Fragment f = nt*4 + kt covers cols [nt*16,nt*16+16), K-slice
// [kt*32, kt*32+32). Per-lane layout (B 32x16 f16): lanes 0-15 col=lane,
// halves j<8 = K kbase+j, j>=8 = K kbase+16+(j-8) with kbase=kt*32;
// lanes 16-31 col=lane-16, same with kbase+8.
// ---------------------------------------------------------------------------
__global__ void pack_weights_kernel(const float* __restrict__ Wq,
                                    const float* __restrict__ Wk,
                                    const float* __restrict__ Wv,
                                    const float* __restrict__ Ws,
                                    _Float16* __restrict__ wpack) {
    int idx = blockIdx.x * blockDim.x + threadIdx.x;   // 0..65535
    int frag   = idx >> 9;          // 0..127
    int within = idx & 511;
    int lane   = within >> 4;
    int j      = within & 15;
    int nt = frag >> 2;
    int kt = frag & 3;
    int col = nt * 16 + (lane & 15);
    int k   = kt * 32 + ((lane & 16) ? 8 : 0) + (j & 7) + ((j & 8) ? 16 : 0);
    int which = col >> 7;
    int c     = col & 127;
    const float* W = (which == 0) ? Wq : (which == 1) ? Wk : (which == 2) ? Wv : Ws;
    wpack[idx] = (_Float16)W[k * HD + c];
}

// ---------------------------------------------------------------------------
// Fused QKVS GEMM: qkvs[n,512] = f16(feat[n,128]) @ f16(Wcat[128,512]) + bias
// One wave per (16-row tile, 128-col group). 8 f32 accum tiles, 4 k-steps.
// ---------------------------------------------------------------------------
__global__ void gemm_qkvs_kernel(const float* __restrict__ feat,
                                 const _Float16* __restrict__ wpack,
                                 const float* __restrict__ bq,
                                 const float* __restrict__ bk,
                                 const float* __restrict__ bv,
                                 const float* __restrict__ bs,
                                 float* __restrict__ qkvs,
                                 int n, int rowTiles) {
    int wave = blockIdx.x * (blockDim.x >> 5) + (threadIdx.x >> 5);
    int lane = threadIdx.x & 31;
    int rt = wave >> 2;
    int cg = wave & 3;
    if (rt >= rowTiles) return;

    v8f acc[8];
    #pragma unroll
    for (int i = 0; i < 8; ++i)
        #pragma unroll
        for (int j = 0; j < 8; ++j) acc[i][j] = 0.0f;

    int m    = lane & 15;
    int row  = rt * 16 + m;
    int rowc = row < n ? row : n - 1;
    const v16h* wv = (const v16h*)wpack;

    #pragma unroll
    for (int kt = 0; kt < 4; ++kt) {
        // A fragment: 16x32 f16, this lane's K base
        const float* ap = feat + (size_t)rowc * IN_F + kt * 32 + ((lane & 16) ? 8 : 0);
        float4 f0 = *(const float4*)(ap + 0);
        float4 f1 = *(const float4*)(ap + 4);
        float4 f2 = *(const float4*)(ap + 16);
        float4 f3 = *(const float4*)(ap + 20);
        v16h A;
        A[0]  = (_Float16)f0.x; A[1]  = (_Float16)f0.y; A[2]  = (_Float16)f0.z; A[3]  = (_Float16)f0.w;
        A[4]  = (_Float16)f1.x; A[5]  = (_Float16)f1.y; A[6]  = (_Float16)f1.z; A[7]  = (_Float16)f1.w;
        A[8]  = (_Float16)f2.x; A[9]  = (_Float16)f2.y; A[10] = (_Float16)f2.z; A[11] = (_Float16)f2.w;
        A[12] = (_Float16)f3.x; A[13] = (_Float16)f3.y; A[14] = (_Float16)f3.z; A[15] = (_Float16)f3.w;

        #pragma unroll
        for (int nt = 0; nt < 8; ++nt) {
            v16h B = wv[(size_t)(((cg * 8 + nt) * 4 + kt)) * 32 + lane];
            acc[nt] = __builtin_amdgcn_wmma_f32_16x16x32_f16(
                false, A, false, B, (short)0, acc[nt], false, false);
        }
    }

    const float* bp = (cg == 0) ? bq : (cg == 1) ? bk : (cg == 2) ? bv : bs;
    int rbase = rt * 16 + ((lane & 16) ? 8 : 0);
    #pragma unroll
    for (int nt = 0; nt < 8; ++nt) {
        int colin = nt * 16 + m;
        int col   = cg * 128 + colin;
        float bias = bp[colin];
        #pragma unroll
        for (int r = 0; r < 8; ++r) {
            int rr = rbase + r;
            if (rr < n) qkvs[(size_t)rr * QKVS_W + col] = acc[nt][r] + bias;
        }
    }
}

// ---------------------------------------------------------------------------
// Edge scores: a[e,h] = dot(q[src],k[dst]) / sqrt(32); fused atomic segment-max.
// One wave per edge; 8 lanes per head (4 elems/lane via float4).
// ---------------------------------------------------------------------------
__global__ void edge_scores_kernel(const float* __restrict__ qkvs,
                                   const int* __restrict__ src,
                                   const int* __restrict__ dst,
                                   float* __restrict__ sc,
                                   unsigned* __restrict__ mU, int e) {
    int wave = blockIdx.x * (blockDim.x >> 5) + (threadIdx.x >> 5);
    if (wave >= e) return;
    int lane = threadIdx.x & 31;
    int s = src[wave], d = dst[wave];
    int h = lane >> 3, sub = lane & 7;
    const float4 qv = *(const float4*)(qkvs + (size_t)s * QKVS_W +       h * 32 + sub * 4);
    const float4 kv = *(const float4*)(qkvs + (size_t)d * QKVS_W + 128 + h * 32 + sub * 4);
    float p = qv.x * kv.x + qv.y * kv.y + qv.z * kv.z + qv.w * kv.w;
    p += __shfl_xor(p, 1); p += __shfl_xor(p, 2); p += __shfl_xor(p, 4);
    if (sub == 0) {
        p *= 0.17677669529663687f;   // 1/sqrt(32)
        sc[(size_t)wave * NH + h] = p;
        atomicMax(mU + (size_t)d * NH + h, fenc(p));
    }
}

// ---------------------------------------------------------------------------
// ex = exp(a - m[dst]); in-place overwrite; atomic segment-sum into den.
// ---------------------------------------------------------------------------
__global__ void exp_sum_kernel(float* __restrict__ sc,
                               const int* __restrict__ dst,
                               const unsigned* __restrict__ mU,
                               float* __restrict__ den, int eh) {
    int i = blockIdx.x * blockDim.x + threadIdx.x;
    if (i >= eh) return;
    int e = i >> 2, h = i & 3;
    int d = dst[e];
    float mv = fdec(mU[(size_t)d * NH + h]);
    float ex = __expf(sc[i] - mv);
    sc[i] = ex;
    atomicAdd(den + (size_t)d * NH + h, ex);
}

// ---------------------------------------------------------------------------
// agg[dst] += (ex/den[dst]) * v[src]  — one wave per edge, 4 cols/lane.
// ---------------------------------------------------------------------------
__global__ void aggregate_kernel(const float* __restrict__ qkvs,
                                 const float* __restrict__ sc,
                                 const float* __restrict__ den,
                                 const int* __restrict__ src,
                                 const int* __restrict__ dst,
                                 float* __restrict__ agg, int e) {
    int wave = blockIdx.x * (blockDim.x >> 5) + (threadIdx.x >> 5);
    if (wave >= e) return;
    int lane = threadIdx.x & 31;
    int s = src[wave], d = dst[wave];
    int h = lane >> 3;
    float sa = sc[(size_t)wave * NH + h] / den[(size_t)d * NH + h];
    const float4 vv = *(const float4*)(qkvs + (size_t)s * QKVS_W + 256 + lane * 4);
    float* ap = agg + (size_t)d * HD + lane * 4;
    atomicAdd(ap + 0, sa * vv.x);
    atomicAdd(ap + 1, sa * vv.y);
    atomicAdd(ap + 2, sa * vv.z);
    atomicAdd(ap + 3, sa * vv.w);
}

// ---------------------------------------------------------------------------
// Gate + LayerNorm + PReLU. One wave per node; 4 cols/lane; wave reductions.
// ---------------------------------------------------------------------------
__global__ void finalize_kernel(const float* __restrict__ qkvs,
                                const float* __restrict__ agg,
                                const float* __restrict__ Wg,
                                const float* __restrict__ bg,
                                const float* __restrict__ gamma,
                                const float* __restrict__ beta,
                                const float* __restrict__ pa,
                                float* __restrict__ out, int n) {
    int wave = blockIdx.x * (blockDim.x >> 5) + (threadIdx.x >> 5);
    if (wave >= n) return;
    int lane = threadIdx.x & 31;
    int c = lane * 4;
    float4 sk = *(const float4*)(qkvs + (size_t)wave * QKVS_W + 384 + c);
    float4 rs = *(const float4*)(agg + (size_t)wave * HD + c);
    float4 g0 = *(const float4*)(Wg + c);
    float4 g1 = *(const float4*)(Wg + 128 + c);
    float4 g2 = *(const float4*)(Wg + 256 + c);
    float z = sk.x * g0.x + sk.y * g0.y + sk.z * g0.z + sk.w * g0.w
            + rs.x * g1.x + rs.y * g1.y + rs.z * g1.z + rs.w * g1.w
            + (sk.x - rs.x) * g2.x + (sk.y - rs.y) * g2.y
            + (sk.z - rs.z) * g2.z + (sk.w - rs.w) * g2.w;
    z += __shfl_xor(z, 1); z += __shfl_xor(z, 2); z += __shfl_xor(z, 4);
    z += __shfl_xor(z, 8); z += __shfl_xor(z, 16);
    float g = 1.0f / (1.0f + __expf(-(z + bg[0])));

    float4 r;
    r.x = g * sk.x + (1.0f - g) * rs.x;
    r.y = g * sk.y + (1.0f - g) * rs.y;
    r.z = g * sk.z + (1.0f - g) * rs.z;
    r.w = g * sk.w + (1.0f - g) * rs.w;

    float s = r.x + r.y + r.z + r.w;
    s += __shfl_xor(s, 1); s += __shfl_xor(s, 2); s += __shfl_xor(s, 4);
    s += __shfl_xor(s, 8); s += __shfl_xor(s, 16);
    float mu = s * (1.0f / 128.0f);

    float d0 = r.x - mu, d1 = r.y - mu, d2 = r.z - mu, d3 = r.w - mu;
    float vs = d0 * d0 + d1 * d1 + d2 * d2 + d3 * d3;
    vs += __shfl_xor(vs, 1); vs += __shfl_xor(vs, 2); vs += __shfl_xor(vs, 4);
    vs += __shfl_xor(vs, 8); vs += __shfl_xor(vs, 16);
    float inv = rsqrtf(vs * (1.0f / 128.0f) + 1e-5f);

    float4 ga = *(const float4*)(gamma + c);
    float4 be = *(const float4*)(beta + c);
    float aslope = pa[0];
    float4 o;
    o.x = d0 * inv * ga.x + be.x;
    o.y = d1 * inv * ga.y + be.y;
    o.z = d2 * inv * ga.z + be.z;
    o.w = d3 * inv * ga.w + be.w;
    o.x = o.x > 0.0f ? o.x : aslope * o.x;
    o.y = o.y > 0.0f ? o.y : aslope * o.y;
    o.z = o.z > 0.0f ? o.z : aslope * o.z;
    o.w = o.w > 0.0f ? o.w : aslope * o.w;
    *(float4*)(out + (size_t)wave * HD + c) = o;
}

// ---------------------------------------------------------------------------
extern "C" void kernel_launch(void* const* d_in, const int* in_sizes, int n_in,
                              void* d_out, int out_size, void* d_ws, size_t ws_size,
                              hipStream_t stream) {
    const float* feat  = (const float*)d_in[0];
    const int*   src   = (const int*)  d_in[1];
    const int*   dst   = (const int*)  d_in[2];
    const float* Wq    = (const float*)d_in[3];
    const float* bq    = (const float*)d_in[4];
    const float* Wk    = (const float*)d_in[5];
    const float* bk    = (const float*)d_in[6];
    const float* Wv    = (const float*)d_in[7];
    const float* bv    = (const float*)d_in[8];
    const float* Ws    = (const float*)d_in[9];
    const float* bs    = (const float*)d_in[10];
    const float* Wg    = (const float*)d_in[11];
    const float* bg    = (const float*)d_in[12];
    const float* gamma = (const float*)d_in[13];
    const float* beta  = (const float*)d_in[14];
    const float* pa    = (const float*)d_in[15];

    const int n = in_sizes[0] / IN_F;
    const int e = in_sizes[1];

    // ---- carve workspace (256B aligned sub-buffers) ----
    char*  ws  = (char*)d_ws;
    size_t off = 0;
    auto carve = [&](size_t bytes) -> char* {
        char* p = ws + off;
        off = (off + bytes + 255) & ~(size_t)255;
        return p;
    };
    float*    qkvs  = (float*)   carve((size_t)n * QKVS_W * sizeof(float));
    _Float16* wpack = (_Float16*)carve((size_t)128 * 512 * sizeof(_Float16));
    float*    sc    = (float*)   carve((size_t)e * NH * sizeof(float));
    char*     zbase = ws + off;           // contiguous zero-init region
    unsigned* mU    = (unsigned*)carve((size_t)n * NH * sizeof(unsigned));
    float*    den   = (float*)   carve((size_t)n * NH * sizeof(float));
    float*    agg   = (float*)   carve((size_t)n * HD * sizeof(float));
    size_t zbytes = (size_t)((ws + off) - zbase);

    hipMemsetAsync(zbase, 0, zbytes, stream);

    pack_weights_kernel<<<65536 / 256, 256, 0, stream>>>(Wq, Wk, Wv, Ws, wpack);

    int rowTiles = (n + 15) / 16;
    int waves    = rowTiles * 4;
    gemm_qkvs_kernel<<<(waves + 7) / 8, 256, 0, stream>>>(
        feat, wpack, bq, bk, bv, bs, qkvs, n, rowTiles);

    edge_scores_kernel<<<(e + 7) / 8, 256, 0, stream>>>(qkvs, src, dst, sc, mU, e);

    int eh = e * NH;
    exp_sum_kernel<<<(eh + 255) / 256, 256, 0, stream>>>(sc, dst, mU, den, eh);

    aggregate_kernel<<<(e + 7) / 8, 256, 0, stream>>>(qkvs, sc, den, src, dst, agg, e);

    finalize_kernel<<<(n + 7) / 8, 256, 0, stream>>>(
        qkvs, agg, Wg, bg, gamma, beta, pa, (float*)d_out, n);
}
